// MultiCodebook_25323127177631
// MI455X (gfx1250) — compile-verified, matrix operand order checked by reading, not run on hardware
//
#include <hip/hip_runtime.h>
#include <hip/hip_bf16.h>

typedef __attribute__((ext_vector_type(16))) _Float16 v16h;
typedef __attribute__((ext_vector_type(8)))  _Float16 v8h;
typedef __attribute__((ext_vector_type(8)))  float    v8f;

#define NUM_EMB 1024
#define CHUNK   64     // embeddings staged into LDS per iteration

// ---------------------------------------------------------------------------
// |e|^2 per embedding row -> workspace
// ---------------------------------------------------------------------------
__global__ void vq_enorm_kernel(const float* __restrict__ emb,
                                float* __restrict__ enorm, int D) {
    int e = blockIdx.x * blockDim.x + threadIdx.x;
    if (e < NUM_EMB) {
        const float* p = emb + (size_t)e * D;
        float s = 0.f;
        for (int d = 0; d < D; ++d) { float v = p[d]; s += v * v; }
        enorm[e] = s;
    }
}

__global__ void vq_zero_kernel(float* p) { p[0] = 0.f; }

// ---------------------------------------------------------------------------
// One VQ level: nearest-codebook search via f16 WMMA (f32 accumulate),
// gather + NCHW store, commitment-loss accumulation.
//   x:    [B, D, H, W] f32      emb: [1024, D] f32
//   qout: [B, D, H, W] f32      lossOut: scalar (atomic)
// Block = 256 threads (8 waves); each wave owns a 16-row tile; block = 128 rows.
// Embedding chunks are double-buffered in LDS so the global->LDS fill of
// chunk c+1 overlaps the WMMA sweep over chunk c (one barrier per chunk).
// ---------------------------------------------------------------------------
template<int D, int HW>
__global__ __launch_bounds__(256)
void vq_level_kernel(const float* __restrict__ x,
                     const float* __restrict__ emb,
                     const float* __restrict__ enorm,
                     float* __restrict__ qout,
                     float* __restrict__ lossOut,
                     float lossScale /* 0.25 / (Nrows*D) */) {
    constexpr int KC   = D / 32;     // WMMA k-chunks per dot product
    constexpr int LSTR = D + 8;      // LDS row stride (halves), keeps 16B align
    __shared__ __align__(16) _Float16 lds_emb[2][CHUNK * LSTR];
    __shared__ int lds_idx[128];

    const int tid  = threadIdx.x;
    const int wave = tid >> 5;
    const int lane = tid & 31;
    const int l15  = lane & 15;
    const int hi8  = (lane >> 4) << 3;           // 0 for lanes 0-15, 8 for 16-31
    const int rowBase = blockIdx.x * 128;
    const int wrow    = rowBase + wave * 16;

    // ---- A fragments: 16 rows x D, f32 -> f16, ISA 16-bit A layout ----
    v16h a[KC];
    float sumsq = 0.f;
    {
        const int row = wrow + l15;              // row this lane supplies
        const int b   = row / HW;
        const int hw  = row - b * HW;
        const float* xb = x + (size_t)b * D * HW + hw;
        #pragma unroll
        for (int kc = 0; kc < KC; ++kc) {
            #pragma unroll
            for (int j = 0; j < 16; ++j) {
                int d = kc * 32 + hi8 + (j < 8 ? j : j + 8);
                float v = xb[(size_t)d * HW];
                a[kc][j] = (_Float16)v;
                sumsq += v * v;
            }
        }
    }
    sumsq += __shfl_xor(sumsq, 16, 32);          // full |x|^2 of row wrow+l15
    float xn[8];
    #pragma unroll
    for (int r = 0; r < 8; ++r) xn[r] = __shfl(sumsq, r + hi8, 32);

    float bestd[8];
    int   besti[8];
    #pragma unroll
    for (int r = 0; r < 8; ++r) { bestd[r] = 3.4e38f; besti[r] = 0; }

    // cooperative f32->f16 stage of one 64-embedding chunk into buffer `buf`
    auto stage = [&](int c, int buf) {
        for (int i = tid; i < CHUNK * D; i += 256) {
            int e = i / D, d = i - e * D;
            lds_emb[buf][e * LSTR + d] =
                (_Float16)emb[(size_t)(c * CHUNK + e) * D + d];
        }
    };

    stage(0, 0);
    __syncthreads();

    // ---- scan the 1024 codes, double-buffered chunks of 64 ----
    for (int c = 0; c < NUM_EMB / CHUNK; ++c) {
        const int cur = c & 1;
        if (c + 1 < NUM_EMB / CHUNK) stage(c + 1, cur ^ 1);   // overlaps compute

        #pragma unroll
        for (int sub = 0; sub < CHUNK / 16; ++sub) {
            v8f acc = {};
            const int erow = sub * 16 + l15;     // embedding (column) this lane supplies
            #pragma unroll
            for (int kc = 0; kc < KC; ++kc) {
                union { v16h v; v8h h[2]; } bf;
                bf.h[0] = *(const v8h*)&lds_emb[cur][erow * LSTR + kc * 32 + hi8];
                bf.h[1] = *(const v8h*)&lds_emb[cur][erow * LSTR + kc * 32 + hi8 + 16];
                acc = __builtin_amdgcn_wmma_f32_16x16x32_f16(
                          false, a[kc], false, bf.v, (short)0, acc, false, false);
            }
            const int   n  = c * CHUNK + sub * 16 + l15;
            const float en = enorm[n];
            #pragma unroll
            for (int r = 0; r < 8; ++r) {
                float dist = xn[r] + en - 2.0f * acc[r];   // == |x-e|^2
                if (dist < bestd[r]) { bestd[r] = dist; besti[r] = n; }
            }
        }
        __syncthreads();   // next chunk's reads + next-next staging both fenced
    }

    // ---- argmin butterfly across the 16 lanes holding each row ----
    #pragma unroll
    for (int off = 1; off < 16; off <<= 1) {
        #pragma unroll
        for (int r = 0; r < 8; ++r) {
            float od = __shfl_xor(bestd[r], off, 32);
            int   oi = __shfl_xor(besti[r], off, 32);
            if (od < bestd[r] || (od == bestd[r] && oi < besti[r])) {
                bestd[r] = od; besti[r] = oi;
            }
        }
    }

    // lanes 0 / 16 publish indices; bestd is exactly |x - e*|^2 -> loss
    if (l15 == 0) {
        float ls = 0.f;
        #pragma unroll
        for (int r = 0; r < 8; ++r) {
            lds_idx[wave * 16 + hi8 + r] = besti[r];
            ls += bestd[r];
        }
        atomicAdd(lossOut, ls * lossScale);
    }
    __syncthreads();

    // ---- gather emb[idx] and store NCHW, coalesced along hw ----
    const int b   = rowBase / HW;                // block never straddles a batch
    const int hw0 = rowBase - b * HW;
    for (int i = tid; i < 128 * D; i += 256) {
        int d = i >> 7;                          // i / 128
        int r = i & 127;                         // row within block
        int k = lds_idx[r];
        qout[(size_t)(b * D + d) * HW + hw0 + r] = emb[(size_t)k * D + d];
    }
}

// ---------------------------------------------------------------------------
// dims: level0 D=64  HW=96*320=30720  rows=245760
//       level1 D=128 HW=48*160=7680   rows=61440
//       level2 D=256 HW=24*80=1920    rows=15360
// d_in (setup_inputs dict order): x0, emb0, x1, emb1, x2, emb2
// d_out: [loss(1) | q0(15728640) | q1(7864320) | q2(3932160)]
// ---------------------------------------------------------------------------
extern "C" void kernel_launch(void* const* d_in, const int* in_sizes, int n_in,
                              void* d_out, int out_size, void* d_ws, size_t ws_size,
                              hipStream_t stream) {
    const float* x0   = (const float*)d_in[0];
    const float* emb0 = (const float*)d_in[1];
    const float* x1   = (const float*)d_in[2];
    const float* emb1 = (const float*)d_in[3];
    const float* x2   = (const float*)d_in[4];
    const float* emb2 = (const float*)d_in[5];
    float* out = (float*)d_out;
    float* en  = (float*)d_ws;               // 3 x 1024 floats

    vq_zero_kernel<<<1, 1, 0, stream>>>(out);

    vq_enorm_kernel<<<NUM_EMB / 256, 256, 0, stream>>>(emb0, en + 0,    64);
    vq_enorm_kernel<<<NUM_EMB / 256, 256, 0, stream>>>(emb1, en + 1024, 128);
    vq_enorm_kernel<<<NUM_EMB / 256, 256, 0, stream>>>(emb2, en + 2048, 256);

    float* q0 = out + 1;
    float* q1 = q0 + 8 * 64 * 30720;   // 15728640
    float* q2 = q1 + 8 * 128 * 7680;   // 7864320

    vq_level_kernel<64, 30720><<<245760 / 128, 256, 0, stream>>>(
        x0, emb0, en + 0, q0, out, 0.25f / (245760.0f * 64.0f));
    vq_level_kernel<128, 7680><<<61440 / 128, 256, 0, stream>>>(
        x1, emb1, en + 1024, q1, out, 0.25f / (61440.0f * 128.0f));
    vq_level_kernel<256, 1920><<<15360 / 128, 256, 0, stream>>>(
        x2, emb2, en + 2048, q2, out, 0.25f / (15360.0f * 256.0f));
}
